// GATLayer_90056874262575
// MI455X (gfx1250) — matrix operand order repaired and checked by reading, not compiled, not run on hardware
//
#include <hip/hip_runtime.h>

#define N_NODES 8192
#define IN_F    512
#define OUT_F   256
#define ALPHA   0.2f
#define NEG_BIG -9e15f

typedef _Float16 v16h __attribute__((ext_vector_type(16)));
typedef _Float16 v8h  __attribute__((ext_vector_type(8)));
typedef float    v8f  __attribute__((ext_vector_type(8)));

// ---------------------------------------------------------------------------
// Kernel 0: W (512x256 f32, row-major) -> WT (256x512 f16): WT[c][k] = W[k][c]
// ---------------------------------------------------------------------------
__global__ void __launch_bounds__(256)
wt_kernel(const float* __restrict__ W, _Float16* __restrict__ WT) {
    int idx = blockIdx.x * 256 + threadIdx.x;   // 512 blocks * 256 = 131072
    int k = idx >> 8;          // 0..511
    int c = idx & 255;         // 0..255
    WT[(size_t)c * IN_F + k] = (_Float16)W[(size_t)k * OUT_F + c];
}

// ---------------------------------------------------------------------------
// Kernel 1: h = x @ W via WMMA f16->f32.
//   Per block: 16 rows x 256 cols (8 waves x 32 cols).
//   Outputs: hhT[col][row] (f16, 256x8192), s[row]=h.a_self, t[row]=h.a_neighs
// ---------------------------------------------------------------------------
__global__ void __launch_bounds__(256)
hproj_kernel(const float* __restrict__ x, const _Float16* __restrict__ WT,
             const float* __restrict__ a_self, const float* __restrict__ a_neighs,
             _Float16* __restrict__ hhT, float* __restrict__ sbuf,
             float* __restrict__ tbuf) {
    __shared__ float spart[8][16];
    __shared__ float tpart[8][16];

    const int tid  = threadIdx.x;
    const int wave = tid >> 5;
    const int lane = tid & 31;
    const int ml   = lane & 15;
    const int half = lane >> 4;
    const int row0 = blockIdx.x * 16;
    const int colbase = wave * 32;

    v8f c0 = {}, c1 = {};
    const float*    xr = x  + (size_t)(row0 + ml) * IN_F;
    const _Float16* w0 = WT + (size_t)(colbase + ml) * IN_F + half * 16;
    const _Float16* w1 = WT + (size_t)(colbase + 16 + ml) * IN_F + half * 16;

    for (int kb = 0; kb < IN_F / 32; ++kb) {
        // A fragment: row ml, K-chunk [half*16,+16) of this 32-slab
        const float* xp = xr + kb * 32 + half * 16;
        v16h a;
        #pragma unroll
        for (int j = 0; j < 8; ++j) {
            a[j]     = (_Float16)xp[j];
            a[8 + j] = (_Float16)xp[8 + j];
        }
        // B fragments: col fixed per lane, 16 contiguous K values
        v16h b0 = *(const v16h*)(w0 + kb * 32);
        v16h b1 = *(const v16h*)(w1 + kb * 32);
        c0 = __builtin_amdgcn_wmma_f32_16x16x32_f16(false, a, false, b0, (short)0, c0, false, false);
        c1 = __builtin_amdgcn_wmma_f32_16x16x32_f16(false, a, false, b1, (short)0, c1, false, false);
    }

    // C layout: lane -> col n = ml, VGPR v -> row M = half*8 + v
    const int col0 = colbase + ml;
    const int col1 = colbase + 16 + ml;
    v8h h0, h1;
    #pragma unroll
    for (int v = 0; v < 8; ++v) { h0[v] = (_Float16)c0[v]; h1[v] = (_Float16)c1[v]; }
    *(v8h*)(hhT + (size_t)col0 * N_NODES + row0 + half * 8) = h0;
    *(v8h*)(hhT + (size_t)col1 * N_NODES + row0 + half * 8) = h1;

    // per-row dot with a_self / a_neighs; reduce over the 16 lanes of each half
    const float as0 = a_self[col0],   as1 = a_self[col1];
    const float an0 = a_neighs[col0], an1 = a_neighs[col1];
    float sv[8], tv[8];
    #pragma unroll
    for (int v = 0; v < 8; ++v) {
        sv[v] = c0[v] * as0 + c1[v] * as1;
        tv[v] = c0[v] * an0 + c1[v] * an1;
    }
    #pragma unroll
    for (int mask = 1; mask < 16; mask <<= 1) {
        #pragma unroll
        for (int v = 0; v < 8; ++v) {
            sv[v] += __shfl_xor(sv[v], mask, 16);
            tv[v] += __shfl_xor(tv[v], mask, 16);
        }
    }
    if (ml == 0) {
        #pragma unroll
        for (int v = 0; v < 8; ++v) {
            spart[wave][half * 8 + v] = sv[v];
            tpart[wave][half * 8 + v] = tv[v];
        }
    }
    __syncthreads();
    if (tid < 16) {
        float ss = 0.f, tt = 0.f;
        #pragma unroll
        for (int w = 0; w < 8; ++w) { ss += spart[w][tid]; tt += tpart[w][tid]; }
        sbuf[row0 + tid] = ss;
        tbuf[row0 + tid] = tt;
    }
}

// ---------------------------------------------------------------------------
// Kernel 2: fused masked-leakyrelu scores -> online softmax -> P@h -> ELU.
//   Per block: 16 output rows, full j-sweep in tiles of 256.
//   Softmax role: thread = (r = tid/16, q = tid%16), 16 cols per thread.
//   PV role: wave w handles output cols [32w, 32w+32) with 2 WMMA C tiles.
// ---------------------------------------------------------------------------
__global__ void __launch_bounds__(256)
gat_attn_kernel(const float* __restrict__ adj, const float* __restrict__ Mmat,
                const float* __restrict__ sbuf, const float* __restrict__ tbuf,
                const _Float16* __restrict__ hhT, float* __restrict__ out) {
    __shared__ __align__(16) _Float16 Pl[16 * 256];   // 8KB probs tile (WMMA A source)
    __shared__ float scaleS[16];
    __shared__ float lsumS[16];

    const int tid  = threadIdx.x;
    const int r    = tid >> 4;     // 0..15 row in block
    const int q    = tid & 15;     // 0..15 col group
    const int wave = tid >> 5;
    const int lane = tid & 31;
    const int ml   = lane & 15;
    const int half = lane >> 4;
    const int row0  = blockIdx.x * 16;
    const int nbase = wave * 32;

    const float  s_r  = sbuf[row0 + r];
    const float* Mrow = Mmat + (size_t)(row0 + r) * N_NODES + q * 16;
    const float* Arow = adj  + (size_t)(row0 + r) * N_NODES + q * 16;
    const _Float16* hp0 = hhT + (size_t)(nbase + ml) * N_NODES + half * 16;
    const _Float16* hp1 = hhT + (size_t)(nbase + 16 + ml) * N_NODES + half * 16;

    float m_run = -__builtin_inff();
    float l_run = 0.f;
    v8f acc0 = {}, acc1 = {};

    for (int jt = 0; jt < N_NODES / 256; ++jt) {
        const int j0 = jt * 256;

        // ---- phase 1: scores + online-softmax stats (registers only) ----
        float e[16];
        float lm = -__builtin_inff();
        #pragma unroll
        for (int i4 = 0; i4 < 4; ++i4) {
            float4 mv = *(const float4*)(Mrow + j0 + i4 * 4);
            float4 av = *(const float4*)(Arow + j0 + i4 * 4);
            float4 tq = *(const float4*)(tbuf + j0 + q * 16 + i4 * 4);
            const float mvv[4] = {mv.x, mv.y, mv.z, mv.w};
            const float avv[4] = {av.x, av.y, av.z, av.w};
            const float tvv[4] = {tq.x, tq.y, tq.z, tq.w};
            #pragma unroll
            for (int k = 0; k < 4; ++k) {
                float val = (s_r + tvv[k]) * mvv[k];
                float ee  = val > 0.f ? val : ALPHA * val;   // leaky relu
                ee = avv[k] > 0.f ? ee : NEG_BIG;            // adjacency mask
                e[i4 * 4 + k] = ee;
                lm = fmaxf(lm, ee);
            }
        }
        #pragma unroll
        for (int mask = 1; mask < 16; mask <<= 1)
            lm = fmaxf(lm, __shfl_xor(lm, mask, 16));
        const float m_new = fmaxf(m_run, lm);
        const float scl   = __expf(m_run - m_new);           // 0 on first tile
        float lsum = 0.f;
        #pragma unroll
        for (int i = 0; i < 16; ++i) {
            float pv = __expf(e[i] - m_new);
            e[i] = pv;
            lsum += pv;
        }
        #pragma unroll
        for (int mask = 1; mask < 16; mask <<= 1)
            lsum += __shfl_xor(lsum, mask, 16);
        l_run = l_run * scl + lsum;
        m_run = m_new;

        __syncthreads();   // previous iteration's WMMA reads of Pl are done
        v8h p0v, p1v;
        #pragma unroll
        for (int i = 0; i < 8; ++i) { p0v[i] = (_Float16)e[i]; p1v[i] = (_Float16)e[8 + i]; }
        *(v8h*)(Pl + r * 256 + q * 16)     = p0v;
        *(v8h*)(Pl + r * 256 + q * 16 + 8) = p1v;
        if (q == 0) scaleS[r] = scl;
        __syncthreads();   // Pl + scaleS visible

        // ---- phase 2: rescale accumulators, P @ h via WMMA ----
        #pragma unroll
        for (int v = 0; v < 8; ++v) {
            const float sc = scaleS[half * 8 + v];
            acc0[v] *= sc;
            acc1[v] *= sc;
        }
        #pragma unroll
        for (int kb = 0; kb < 8; ++kb) {
            union { v16h v; v8h h[2]; } a;
            a.h[0] = *(const v8h*)(Pl + ml * 256 + kb * 32 + half * 16);
            a.h[1] = *(const v8h*)(Pl + ml * 256 + kb * 32 + half * 16 + 8);
            v16h b0 = *(const v16h*)(hp0 + j0 + kb * 32);
            v16h b1 = *(const v16h*)(hp1 + j0 + kb * 32);
            acc0 = __builtin_amdgcn_wmma_f32_16x16x32_f16(false, a.v, false, b0, (short)0, acc0, false, false);
            acc1 = __builtin_amdgcn_wmma_f32_16x16x32_f16(false, a.v, false, b1, (short)0, acc1, false, false);
        }
    }

    if (q == 0) lsumS[r] = l_run;
    __syncthreads();

    // normalize, ELU, store (C layout: col n = ml, VGPR v -> row M = half*8+v)
    #pragma unroll
    for (int v = 0; v < 8; ++v) {
        const int rowM = half * 8 + v;
        const float inv = 1.f / lsumS[rowM];
        float x0 = acc0[v] * inv;
        float x1 = acc1[v] * inv;
        x0 = x0 > 0.f ? x0 : __expf(x0) - 1.f;
        x1 = x1 > 0.f ? x1 : __expf(x1) - 1.f;
        out[(size_t)(row0 + rowM) * OUT_F + nbase + ml]      = x0;
        out[(size_t)(row0 + rowM) * OUT_F + nbase + 16 + ml] = x1;
    }
}

// ---------------------------------------------------------------------------
extern "C" void kernel_launch(void* const* d_in, const int* in_sizes, int n_in,
                              void* d_out, int out_size, void* d_ws, size_t ws_size,
                              hipStream_t stream) {
    const float* x        = (const float*)d_in[0];
    const float* adj      = (const float*)d_in[1];
    const float* Mm       = (const float*)d_in[2];
    const float* W        = (const float*)d_in[3];
    const float* a_self   = (const float*)d_in[4];
    const float* a_neighs = (const float*)d_in[5];
    float* out = (float*)d_out;

    char* ws = (char*)d_ws;
    _Float16* WT  = (_Float16*)ws;                                   // 256KB
    _Float16* hhT = (_Float16*)(ws + (size_t)IN_F * OUT_F * 2);      // 4MB
    float* sbuf = (float*)(ws + (size_t)IN_F * OUT_F * 2 + (size_t)OUT_F * N_NODES * 2);
    float* tbuf = sbuf + N_NODES;

    wt_kernel<<<512, 256, 0, stream>>>(W, WT);
    hproj_kernel<<<512, 256, 0, stream>>>(x, WT, a_self, a_neighs, hhT, sbuf, tbuf);
    gat_attn_kernel<<<N_NODES / 16, 256, 0, stream>>>(adj, Mm, sbuf, tbuf, hhT, out);
}
